// NonLocal_34299608826352
// MI455X (gfx1250) — matrix operand order repaired and checked by reading, not compiled, not run on hardware
//
#include <hip/hip_runtime.h>
#include <hip/hip_bf16.h>

// ---------------------------------------------------------------------------
// NonLocal attention block, fused flash-style for MI455X (gfx1250, wave32).
// B=8, C=256, Ci=32, N=64*64=4096.
//
// Never materializes the [B,4096,4096] attention matrix (512MB fp32 -> ~90us
// of HBM traffic at 23.3 TB/s). Instead:
//   stage_a : y1,y2,y3 = silu(bn(w @ x)), stored bf16 in WMMA-friendly layouts
//   stage_b : Z[n] = sum_m exp(c[n,m]) via v_wmma_f32_16x16x32_bf16 (K=Ci=32)
//   stage_c : out[m,c] += wmma( bf16(exp(c_tile)*Zinv) , y3_tile ) -- the
//             D-layout -> A-layout transpose is lane-local (register packs only)
//   stage_d : mask = silu(bn(w4 @ y_last)) + x
// ---------------------------------------------------------------------------

#define BATCH 8
#define CIN   256
#define CI    32
#define NPIX  4096

typedef __attribute__((ext_vector_type(16))) __bf16 v16bf;
typedef __attribute__((ext_vector_type(8)))  float  v8f;

union FragBF {
  v16bf v;
  uint4 u[2];
  __bf16 e[16];
};
union FragF {
  v8f  v;
  float f[8];
};

// workspace layout (bytes)
#define OFF_Y1T   0u                        // bf16 [B][N][CI]  : 2 MB
#define OFF_Y2T   (2u*1024u*1024u)          // bf16 [B][N][CI]  : 2 MB
#define OFF_Y3C   (4u*1024u*1024u)          // bf16 [B][CI][N]  : 2 MB
#define OFF_ZINV  (6u*1024u*1024u)          // f32  [B][N]      : 128 KB
#define OFF_YLAST (6u*1024u*1024u + 256u*1024u) // f32 [B][N][CI]: 4 MB

// ---------------------------------------------------------------------------
// Stage A: y = silu(scale * (w @ x) + bias).  blockIdx.y selects y1/y2/y3.
// Weights are read with wave-uniform indices -> scalar-cache s_loads.
// y1t/y2t: [b][n][k] bf16 (A/B fragment source).  y3c: [b][c][n] bf16.
// ---------------------------------------------------------------------------
__global__ void __launch_bounds__(256) stage_a(
    const float* __restrict__ x,
    const float* __restrict__ w1, const float* __restrict__ s1, const float* __restrict__ c1,
    const float* __restrict__ w2, const float* __restrict__ s2, const float* __restrict__ c2,
    const float* __restrict__ w3, const float* __restrict__ s3, const float* __restrict__ c3,
    __bf16* __restrict__ y1t, __bf16* __restrict__ y2t, __bf16* __restrict__ y3c) {
  const int variant = blockIdx.y;
  const float* w  = (variant == 0) ? w1 : ((variant == 1) ? w2 : w3);
  const float* sc = (variant == 0) ? s1 : ((variant == 1) ? s2 : s3);
  const float* bi = (variant == 0) ? c1 : ((variant == 1) ? c2 : c3);

  const int g = blockIdx.x * 256 + threadIdx.x;   // g = b*N + n
  const int b = g >> 12;
  const int n = g & (NPIX - 1);
  const float* xb = x + (size_t)b * CIN * NPIX + n;

  float acc[CI];
#pragma unroll
  for (int o = 0; o < CI; ++o) acc[o] = 0.f;

  for (int i0 = 0; i0 < CIN; i0 += 8) {
    float xv[8];
#pragma unroll
    for (int k = 0; k < 8; ++k) xv[k] = xb[(size_t)(i0 + k) * NPIX];
#pragma unroll
    for (int o = 0; o < CI; ++o) {
#pragma unroll
      for (int k = 0; k < 8; ++k)
        acc[o] = fmaf(w[o * CIN + i0 + k], xv[k], acc[o]);
    }
  }

  if (variant < 2) {
    union { __bf16 e[CI]; uint4 u[4]; } st;
#pragma unroll
    for (int o = 0; o < CI; ++o) {
      float t = acc[o] * sc[o] + bi[o];
      float sil = t / (1.f + __expf(-t));
      st.e[o] = (__bf16)sil;
    }
    __bf16* dst = ((variant == 0) ? y1t : y2t) + (size_t)g * CI;
    uint4* du = reinterpret_cast<uint4*>(dst);
#pragma unroll
    for (int q = 0; q < 4; ++q) du[q] = st.u[q];
  } else {
#pragma unroll
    for (int o = 0; o < CI; ++o) {
      float t = acc[o] * sc[o] + bi[o];
      float sil = t / (1.f + __expf(-t));
      y3c[((size_t)b * CI + o) * NPIX + n] = (__bf16)sil;
    }
  }
}

// ---------------------------------------------------------------------------
// Stage B: Z^-1[n] = 1 / sum_m exp(c[n,m]),  c = y2^T y1 (K = Ci = 32).
// One wave owns 16 rows n; loops m over 4096 in 16-wide tiles; one WMMA per
// tile; per-lane partial exp-sums, cross-lane reduce in 16-lane halves at end.
// ---------------------------------------------------------------------------
__global__ void __launch_bounds__(256) stage_b(
    const __bf16* __restrict__ y1t, const __bf16* __restrict__ y2t,
    float* __restrict__ zinv) {
  const int lane = threadIdx.x & 31;
  const int wid  = threadIdx.x >> 5;
  const int b    = blockIdx.y;
  const int n0   = (blockIdx.x * 8 + wid) * 16;
  const int lrow = lane & 15;
  const int hi   = lane >> 4;

  // A fragment: y2^T rows n0..n0+15.  Lane layout per ISA 16-bit A 16x32:
  // lanes 0-15 hold K 0..7 / 16..23, lanes 16-31 hold K 8..15 / 24..31.
  FragBF fa;
  {
    const __bf16* p = y2t + ((size_t)(b * NPIX + n0 + lrow)) * CI + hi * 8;
    fa.u[0] = *reinterpret_cast<const uint4*>(p);
    fa.u[1] = *reinterpret_cast<const uint4*>(p + 16);
  }

  const v8f zc = {0.f, 0.f, 0.f, 0.f, 0.f, 0.f, 0.f, 0.f};
  float psum[8];
#pragma unroll
  for (int v = 0; v < 8; ++v) psum[v] = 0.f;

  // B fragment: lanes 0-15 hold K 0..7 / 16..23, lanes 16-31 hold K 8..15 / 24..31 (column m).
  const __bf16* pb = y1t + ((size_t)(b * NPIX + lrow)) * CI + hi * 8;
  for (int mt = 0; mt < NPIX / 16; ++mt) {
    FragBF fb;
    const __bf16* p = pb + (size_t)mt * 16 * CI;
    fb.u[0] = *reinterpret_cast<const uint4*>(p);
    fb.u[1] = *reinterpret_cast<const uint4*>(p + 16);
    FragF c;
    c.v = __builtin_amdgcn_wmma_f32_16x16x32_bf16(
        false, fa.v, false, fb.v, (short)0, zc, false, false);
#pragma unroll
    for (int v = 0; v < 8; ++v) psum[v] += __expf(c.f[v]);
  }

  // reduce across the 16-lane half (row n stays within a half for masks < 16)
#pragma unroll
  for (int v = 0; v < 8; ++v) {
    psum[v] += __shfl_xor(psum[v], 1);
    psum[v] += __shfl_xor(psum[v], 2);
    psum[v] += __shfl_xor(psum[v], 4);
    psum[v] += __shfl_xor(psum[v], 8);
  }
  if (lrow == 0) {
#pragma unroll
    for (int v = 0; v < 8; ++v)
      zinv[(size_t)b * NPIX + n0 + v + hi * 8] = 1.0f / psum[v];
  }
}

// ---------------------------------------------------------------------------
// Stage C: for each 16-column block m0:
//   loop n in chunks of 32:
//     Da,Db = wmma(y2^T[n..],  y1[..,m])            (two 16x16 c tiles)
//     w     = exp(c) * Zinv[n]  -> bf16, packed lane-locally into A fragment
//     acc  += wmma(w^T[16m x 32n], y3^T[32n x 16c])  (c split 0-15 / 16-31)
// Result stored as y_last [b][m][c] f32 (coalesced per D-register).
// ---------------------------------------------------------------------------
__global__ void __launch_bounds__(256) stage_c(
    const __bf16* __restrict__ y1t, const __bf16* __restrict__ y2t,
    const __bf16* __restrict__ y3c, const float* __restrict__ zinv,
    float* __restrict__ ylast) {
  __shared__ float zsh[NPIX];
  const int tid = threadIdx.x;
  const int b   = blockIdx.y;
  {
    const float4* zp = reinterpret_cast<const float4*>(zinv + (size_t)b * NPIX);
    float4* zs = reinterpret_cast<float4*>(zsh);
#pragma unroll
    for (int q = 0; q < 4; ++q) zs[q * 256 + tid] = zp[q * 256 + tid];
  }
  __syncthreads();

  const int lane = tid & 31;
  const int wid  = tid >> 5;
  const int m0   = (blockIdx.x * 8 + wid) * 16;
  const int lrow = lane & 15;
  const int hi   = lane >> 4;

  // fixed B fragment for the c-tile WMMAs: y1 columns m0..m0+15
  FragBF fbC;
  {
    const __bf16* p = y1t + ((size_t)(b * NPIX + m0 + lrow)) * CI + hi * 8;
    fbC.u[0] = *reinterpret_cast<const uint4*>(p);
    fbC.u[1] = *reinterpret_cast<const uint4*>(p + 16);
  }

  const v8f zc = {0.f, 0.f, 0.f, 0.f, 0.f, 0.f, 0.f, 0.f};
  FragF acc0, acc1;
#pragma unroll
  for (int v = 0; v < 8; ++v) { acc0.f[v] = 0.f; acc1.f[v] = 0.f; }

  const __bf16* pA   = y2t + ((size_t)(b * NPIX) + lrow) * CI + hi * 8;
  const __bf16* p3lo = y3c + ((size_t)(b * CI) + lrow) * NPIX + hi * 8;
  const __bf16* p3hi = p3lo + (size_t)16 * NPIX;

  for (int nc = 0; nc < NPIX / 32; ++nc) {
    const int n0 = nc * 32;
    FragBF falo, fahi;
    const __bf16* pa = pA + (size_t)n0 * CI;
    falo.u[0] = *reinterpret_cast<const uint4*>(pa);
    falo.u[1] = *reinterpret_cast<const uint4*>(pa + 16);
    pa += 16 * CI;
    fahi.u[0] = *reinterpret_cast<const uint4*>(pa);
    fahi.u[1] = *reinterpret_cast<const uint4*>(pa + 16);

    FragF Da, Db;
    Da.v = __builtin_amdgcn_wmma_f32_16x16x32_bf16(
        false, falo.v, false, fbC.v, (short)0, zc, false, false);
    Db.v = __builtin_amdgcn_wmma_f32_16x16x32_bf16(
        false, fahi.v, false, fbC.v, (short)0, zc, false, false);

    // Zinv for the rows held by this lane-half (D reg v <-> rows v, v+8)
    const int zb = n0 + hi * 8;
    float za[8] __attribute__((aligned(16)));
    float zbv[8] __attribute__((aligned(16)));
    reinterpret_cast<float4*>(za)[0]  = *reinterpret_cast<const float4*>(&zsh[zb]);
    reinterpret_cast<float4*>(za)[1]  = *reinterpret_cast<const float4*>(&zsh[zb + 4]);
    reinterpret_cast<float4*>(zbv)[0] = *reinterpret_cast<const float4*>(&zsh[zb + 16]);
    reinterpret_cast<float4*>(zbv)[1] = *reinterpret_cast<const float4*>(&zsh[zb + 20]);

    // lane-local transpose: D-layout (f32) -> A-layout (bf16) of w^T
    FragBF a2;
#pragma unroll
    for (int t = 0; t < 8; ++t) {
      a2.e[t]     = (__bf16)(__expf(Da.f[t]) * za[t]);
      a2.e[8 + t] = (__bf16)(__expf(Db.f[t]) * zbv[t]);
    }

    FragBF b2lo, b2hi;
    b2lo.u[0] = *reinterpret_cast<const uint4*>(p3lo + n0);
    b2lo.u[1] = *reinterpret_cast<const uint4*>(p3lo + n0 + 16);
    b2hi.u[0] = *reinterpret_cast<const uint4*>(p3hi + n0);
    b2hi.u[1] = *reinterpret_cast<const uint4*>(p3hi + n0 + 16);

    acc0.v = __builtin_amdgcn_wmma_f32_16x16x32_bf16(
        false, a2.v, false, b2lo.v, (short)0, acc0.v, false, false);
    acc1.v = __builtin_amdgcn_wmma_f32_16x16x32_bf16(
        false, a2.v, false, b2hi.v, (short)0, acc1.v, false, false);
  }

  // store y_last [b][m][c] f32 — lanes 0-15 and 16-31 each write a contiguous
  // 64B run per D register.
#pragma unroll
  for (int v = 0; v < 8; ++v) {
    const int m = m0 + v + hi * 8;
    float* o = ylast + ((size_t)(b * NPIX + m)) * CI + lrow;
    o[0]  = acc0.f[v];
    o[16] = acc1.f[v];
  }
}

// ---------------------------------------------------------------------------
// Stage D: out = silu(s4 * (w4 @ y_last) + b4) + x.  One thread per (b,m);
// 32 y_last values in registers, 256 output channels, w4 via scalar cache.
// ---------------------------------------------------------------------------
__global__ void __launch_bounds__(256) stage_d(
    const float* __restrict__ ylast, const float* __restrict__ w4,
    const float* __restrict__ s4, const float* __restrict__ b4,
    const float* __restrict__ x, float* __restrict__ out) {
  const int g = blockIdx.x * 256 + threadIdx.x;
  const int b = g >> 12;
  const int m = g & (NPIX - 1);

  float yl[CI] __attribute__((aligned(16)));
  const float4* p = reinterpret_cast<const float4*>(ylast + (size_t)g * CI);
#pragma unroll
  for (int q = 0; q < 8; ++q)
    reinterpret_cast<float4*>(yl)[q] = p[q];

#pragma unroll 4
  for (int co = 0; co < CIN; ++co) {
    float acc = 0.f;
#pragma unroll
    for (int ci = 0; ci < CI; ++ci)
      acc = fmaf(w4[co * CI + ci], yl[ci], acc);
    float t = acc * s4[co] + b4[co];
    float sil = t / (1.f + __expf(-t));
    const size_t idx = ((size_t)(b * CIN + co)) * NPIX + m;
    out[idx] = sil + x[idx];
  }
}

// ---------------------------------------------------------------------------
extern "C" void kernel_launch(void* const* d_in, const int* in_sizes, int n_in,
                              void* d_out, int out_size, void* d_ws, size_t ws_size,
                              hipStream_t stream) {
  const float* x  = (const float*)d_in[0];
  const float* w1 = (const float*)d_in[1];
  const float* s1 = (const float*)d_in[2];
  const float* b1 = (const float*)d_in[3];
  const float* w2 = (const float*)d_in[4];
  const float* s2 = (const float*)d_in[5];
  const float* b2 = (const float*)d_in[6];
  const float* w3 = (const float*)d_in[7];
  const float* s3 = (const float*)d_in[8];
  const float* b3 = (const float*)d_in[9];
  const float* w4 = (const float*)d_in[10];
  const float* s4 = (const float*)d_in[11];
  const float* b4 = (const float*)d_in[12];

  char* ws = (char*)d_ws;
  __bf16* y1t  = (__bf16*)(ws + OFF_Y1T);
  __bf16* y2t  = (__bf16*)(ws + OFF_Y2T);
  __bf16* y3c  = (__bf16*)(ws + OFF_Y3C);
  float*  zinv = (float*)(ws + OFF_ZINV);
  float*  ylast= (float*)(ws + OFF_YLAST);

  // A: 128 blocks x 3 variants (y1/y2/y3)
  stage_a<<<dim3(BATCH * NPIX / 256, 3), 256, 0, stream>>>(
      x, w1, s1, b1, w2, s2, b2, w3, s3, b3, y1t, y2t, y3c);
  // B: 8 waves/block, each wave owns 16 rows n;  32 blocks * 8 waves * 16 = 4096
  stage_b<<<dim3(NPIX / 16 / 8, BATCH), 256, 0, stream>>>(y1t, y2t, zinv);
  // C: same decomposition over m
  stage_c<<<dim3(NPIX / 16 / 8, BATCH), 256, 0, stream>>>(y1t, y2t, y3c, zinv, ylast);
  // D: final projection + residual
  stage_d<<<dim3(BATCH * NPIX / 256), 256, 0, stream>>>(ylast, w4, s4, b4, x,
                                                        (float*)d_out);
}
